// GraphTransformer_56195352101387
// MI455X (gfx1250) — matrix operand order repaired and checked by reading, not compiled, not run on hardware
//
#include <hip/hip_runtime.h>
#include <hip/hip_bf16.h>

// ---------------------------------------------------------------------------
// GraphTransformer forward for MI455X (gfx1250, wave32, WMMA).
// Dense linears: v_wmma_f32_16x16x32_f16.  Weights are pre-converted once to
// transposed f16 (Wt[N][K]) so B fragments are direct 2x global_load_b128
// per lane (cache-resident); A tiles are double-buffered f16 in LDS.
// Graph ops (gather/scatter/softmax/LN) are VALU kernels.
// ---------------------------------------------------------------------------

typedef __attribute__((ext_vector_type(16))) _Float16 v16h;
typedef __attribute__((ext_vector_type(8)))  float    v8f;
typedef __attribute__((ext_vector_type(4)))  float    f32x4;
typedef __attribute__((ext_vector_type(4)))  _Float16 h16x4;
typedef __attribute__((ext_vector_type(8)))  _Float16 h16x8;

#define NNODES 102400
#define NEDGES 409600
#define NB     2048
#define NPG    50
#define SEEDS  75
#define ECHUNK 102400   // 4 chunks of edges

// one-time: Wt[n][k] = (f16) W[k][n]
__global__ void k_cvt_w(const float* __restrict__ w, _Float16* __restrict__ wt,
                        int K, int Ncol)
{
  int i = blockIdx.x * blockDim.x + threadIdx.x;   // over K*Ncol, coalesced read
  if (i >= K * Ncol) return;
  int k = i / Ncol, n = i - k * Ncol;
  wt[(size_t)n * K + k] = (_Float16)w[i];
}

// ---------------------------------------------------------------------------
// GEMM:  C[M,Ncol] = act(A[M,K] @ W[K,Ncol] + bias),  W given as f16 Wt[N][K].
// Block = 256 threads = 8 waves; block tile 64x64; wave tile 32x16 (2 WMMAs
// sharing one B fragment); k-step 32; A tile double-buffered in LDS.
// K multiple of 32, Ncol multiple of 64, M arbitrary. act:0=none,1=silu,2=relu
// ---------------------------------------------------------------------------
#define LPAD 40   // halves per LDS row (80B, multiple of 16B)

__global__ __launch_bounds__(256) void k_gemm(
    const float* __restrict__ A, const _Float16* __restrict__ Wt,
    const float* __restrict__ bias, float* __restrict__ C,
    int M, int K, int Ncol, int ldc, int act)
{
  __shared__ __align__(16) _Float16 As[2][64 * LPAD];   // 2 x (64 rows x 32 k)
  const int bm0 = blockIdx.x * 64;
  const int bn0 = blockIdx.y * 64;
  const int tid = threadIdx.x;
  const int wid = tid >> 5, lane = tid & 31;
  const int wm = wid >> 2, wn = wid & 3;      // wm: 32-row strip, wn: 16-col strip
  const int rr = lane & 15, hi = lane >> 4;

  v8f acc0 = {0.f,0.f,0.f,0.f,0.f,0.f,0.f,0.f};
  v8f acc1 = {0.f,0.f,0.f,0.f,0.f,0.f,0.f,0.f};
  const bool fullM = (bm0 + 64 <= M);

  const int ar = tid >> 3;          // 0..31 : stages rows ar and ar+32
  const int ac = (tid & 7) * 4;     // k offset 0,4,...,28

  auto stage = [&](int kt, int buf) {
    if (fullM) {
      f32x4 v0 = *(const f32x4*)&A[(size_t)(bm0 + ar)      * K + kt + ac];
      f32x4 v1 = *(const f32x4*)&A[(size_t)(bm0 + ar + 32) * K + kt + ac];
      *(h16x4*)&As[buf][ar * LPAD + ac]        = __builtin_convertvector(v0, h16x4);
      *(h16x4*)&As[buf][(ar + 32) * LPAD + ac] = __builtin_convertvector(v1, h16x4);
    } else {
#pragma unroll
      for (int p = 0; p < 2; ++p) {
        int row = bm0 + ar + p * 32;
        f32x4 v = {0.f,0.f,0.f,0.f};
        if (row < M) v = *(const f32x4*)&A[(size_t)row * K + kt + ac];
        *(h16x4*)&As[buf][(ar + p * 32) * LPAD + ac] = __builtin_convertvector(v, h16x4);
      }
    }
  };

  stage(0, 0);
  __syncthreads();

  // B fragment source: row (bn0+wn*16+rr) of Wt, contiguous in k.
  const _Float16* wrow = Wt + (size_t)(bn0 + wn * 16 + rr) * K;
  const int arow0 = (wm * 32 + rr) * LPAD;
  const int arow1 = (wm * 32 + 16 + rr) * LPAD;
  const int nk = K >> 5;

  for (int i = 0; i < nk; ++i) {
    const int buf = i & 1;
    if (i + 1 < nk) stage((i + 1) * 32, buf ^ 1);   // overlap with compute

    union { v16h v; h16x8 h[2]; } a0, a1, b;
    b.h[0] = *(const h16x8*)&wrow[i * 32 + hi * 8];      // global, cache-hot
    b.h[1] = *(const h16x8*)&wrow[i * 32 + 16 + hi * 8];
    a0.h[0] = *(const h16x8*)&As[buf][arow0 + hi * 8];    // 4 x ds_load_b128
    a0.h[1] = *(const h16x8*)&As[buf][arow0 + 16 + hi * 8];
    a1.h[0] = *(const h16x8*)&As[buf][arow1 + hi * 8];
    a1.h[1] = *(const h16x8*)&As[buf][arow1 + 16 + hi * 8];

    acc0 = __builtin_amdgcn_wmma_f32_16x16x32_f16(
        false, a0.v, false, b.v, (short)0, acc0, false, false);
    acc1 = __builtin_amdgcn_wmma_f32_16x16x32_f16(
        false, a1.v, false, b.v, (short)0, acc1, false, false);
    __syncthreads();
  }

  // ---- epilogue: C layout VGPR i -> row_local = 8*hi + i, col = lane&15 ----
  const int col = bn0 + wn * 16 + rr;
  const float bv = bias ? bias[col] : 0.f;
#pragma unroll
  for (int i = 0; i < 8; ++i) {
    int row0 = bm0 + wm * 32 + hi * 8 + i;
    int row1 = row0 + 16;
    float v0 = acc0[i] + bv;
    float v1 = acc1[i] + bv;
    if (act == 1)      { v0 = v0 / (1.f + __expf(-v0)); v1 = v1 / (1.f + __expf(-v1)); }
    else if (act == 2) { v0 = fmaxf(v0, 0.f);           v1 = fmaxf(v1, 0.f); }
    if (row0 < M) C[(size_t)row0 * ldc + col] = v0;
    if (row1 < M) C[(size_t)row1 * ldc + col] = v1;
  }
}

// ---------------------------------------------------------------------------
// LayerNorm over last dim = 64, one wave per row (wave32: 2 elems / lane).
//   t   = x[row] (+ x2[row] if x2)
//   y   = (resid[row] if resid) + LN(t)*g + b
// ---------------------------------------------------------------------------
__global__ __launch_bounds__(256) void k_layernorm(
    const float* __restrict__ x,  int sx,
    const float* __restrict__ x2, int sx2,
    const float* __restrict__ g,  const float* __restrict__ bb,
    const float* __restrict__ resid, int sr,
    float* __restrict__ y, int sy, int rows)
{
  int row  = (blockIdx.x * blockDim.x + threadIdx.x) >> 5;
  int lane = threadIdx.x & 31;
  if (row >= rows) return;
  const float* xr = x + (size_t)row * sx;
  float v0 = xr[lane], v1 = xr[lane + 32];
  if (x2) {
    const float* x2r = x2 + (size_t)row * sx2;
    v0 += x2r[lane]; v1 += x2r[lane + 32];
  }
  float s = v0 + v1, ss = v0 * v0 + v1 * v1;
#pragma unroll
  for (int o = 16; o; o >>= 1) {
    s  += __shfl_xor(s,  o, 32);
    ss += __shfl_xor(ss, o, 32);
  }
  float mean = s * 0.015625f;
  float var  = ss * 0.015625f - mean * mean;
  float inv  = rsqrtf(var + 1e-5f);
  float o0 = (v0 - mean) * inv * g[lane]      + bb[lane];
  float o1 = (v1 - mean) * inv * g[lane + 32] + bb[lane + 32];
  if (resid) {
    o0 += resid[(size_t)row * sr + lane];
    o1 += resid[(size_t)row * sr + lane + 32];
  }
  float* yr = y + (size_t)row * sy;
  yr[lane] = o0; yr[lane + 32] = o1;
}

// ---------------------------------------------------------------------------
__global__ void k_fill(float* p, float v, long n) {
  long i = (long)blockIdx.x * blockDim.x + threadIdx.x;
  if (i < n) p[i] = v;
}

__global__ void k_embed(const int* __restrict__ a0, const int* __restrict__ a1,
                        const int* __restrict__ a2, const int* __restrict__ a3,
                        const float* __restrict__ emb, float* __restrict__ e)
{
  int i = blockIdx.x * blockDim.x + threadIdx.x;      // over N*64
  if (i >= NNODES * 64) return;
  int node = i >> 6, c = i & 63;
  float* row = e + (size_t)node * 256;
  row[c]        = emb[(size_t)a0[node] * 64 + c];
  row[64 + c]   = emb[(size_t)a1[node] * 64 + c];
  row[128 + c]  = emb[(size_t)a2[node] * 64 + c];
  row[192 + c]  = emb[(size_t)a3[node] * 64 + c];
}

// m[le] = concat(f[dst[e]], f[src[e]])   (dst first, per reference)
__global__ void k_edge_concat(const float* __restrict__ f,
                              const int* __restrict__ src, const int* __restrict__ dst,
                              float* __restrict__ m, int e0, int ecount)
{
  int i = blockIdx.x * blockDim.x + threadIdx.x;      // over ecount*128
  if (i >= ecount * 128) return;
  int le = i >> 7, c = i & 127;
  int e = e0 + le;
  int n = (c < 64) ? dst[e] : src[e];
  m[(size_t)le * 128 + c] = f[(size_t)n * 64 + (c & 63)];
}

// out[dst[e]] += m[le]   (64 cols, atomics)
__global__ void k_scatter_add(const float* __restrict__ m,
                              const int* __restrict__ dst,
                              float* __restrict__ out, int e0, int ecount)
{
  int i = blockIdx.x * blockDim.x + threadIdx.x;      // over ecount*64
  if (i >= ecount * 64) return;
  int le = i >> 6, c = i & 63;
  atomicAdd(&out[(size_t)dst[e0 + le] * 64 + c], m[(size_t)le * 64 + c]);
}

__global__ void k_deg_scatter(const int* __restrict__ dst, float* __restrict__ deg) {
  int e = blockIdx.x * blockDim.x + threadIdx.x;
  if (e < NEDGES) atomicAdd(&deg[dst[e]], 1.0f);
}

__global__ void k_rsqrt_inplace(float* p, int n) {
  int i = blockIdx.x * blockDim.x + threadIdx.x;
  if (i < n) p[i] = rsqrtf(p[i]);
}

// agg[dst] += xw[src] * dinv[src]*dinv[dst]
__global__ void k_gcn_scatter(const float* __restrict__ xw,
                              const int* __restrict__ src, const int* __restrict__ dst,
                              const float* __restrict__ dinv, float* __restrict__ agg)
{
  int i = blockIdx.x * blockDim.x + threadIdx.x;      // over E*64
  if (i >= NEDGES * 64) return;
  int e = i >> 6, c = i & 63;
  int s = src[e], d = dst[e];
  atomicAdd(&agg[(size_t)d * 64 + c],
            xw[(size_t)s * 64 + c] * dinv[s] * dinv[d]);
}

// out = agg + xw * dinv^2 + bias   (out may alias agg)
__global__ void k_gcn_finish(const float* agg, const float* __restrict__ xw,
                             const float* __restrict__ dinv,
                             const float* __restrict__ bias, float* out)
{
  int i = blockIdx.x * blockDim.x + threadIdx.x;      // over N*64
  if (i >= NNODES * 64) return;
  int node = i >> 6, c = i & 63;
  out[i] = agg[i] + xw[i] * dinv[node] * dinv[node] + bias[c];
}

// ---------------------------------------------------------------------------
// MAB attention core, one block per graph b:
//   S[i,j] = <q[i], K[b,j]> / 8 ; A = softmax over QUERY axis i (faithful!)
//   out[b,i,:] = q[i,:] + sum_j A[i,j] * V[b,j,:]
// q is shared [75,64] (qPerB=0) or per-graph [B,75,64] (qPerB=1).
// ---------------------------------------------------------------------------
__global__ __launch_bounds__(256) void k_attn(
    const float* __restrict__ qg, int qPerB,
    const float* __restrict__ K, const float* __restrict__ V,
    float* __restrict__ out, int Kr)
{
  __shared__ float sq[SEEDS][64];
  __shared__ float sk[SEEDS][64];
  __shared__ float sS[SEEDS][SEEDS + 1];
  int b = blockIdx.x;
  const float* qb = qPerB ? (qg + (size_t)b * SEEDS * 64) : qg;
  const float* Kb = K + (size_t)b * Kr * 64;
  const float* Vb = V + (size_t)b * Kr * 64;

  for (int t = threadIdx.x; t < SEEDS * 64; t += blockDim.x)
    sq[t >> 6][t & 63] = qb[t];
  for (int t = threadIdx.x; t < Kr * 64; t += blockDim.x)
    sk[t >> 6][t & 63] = Kb[t];
  __syncthreads();

  for (int t = threadIdx.x; t < SEEDS * Kr; t += blockDim.x) {
    int i = t / Kr, j = t - i * Kr;
    float acc = 0.f;
#pragma unroll 8
    for (int d = 0; d < 64; ++d) acc += sq[i][d] * sk[j][d];
    sS[i][j] = acc * 0.125f;
  }
  __syncthreads();

  // softmax over queries i, per column j
  for (int j = threadIdx.x; j < Kr; j += blockDim.x) {
    float mx = -1e30f;
    for (int i = 0; i < SEEDS; ++i) mx = fmaxf(mx, sS[i][j]);
    float sum = 0.f;
    for (int i = 0; i < SEEDS; ++i) {
      float e = __expf(sS[i][j] - mx);
      sS[i][j] = e; sum += e;
    }
    float inv = 1.f / sum;
    for (int i = 0; i < SEEDS; ++i) sS[i][j] *= inv;
  }
  __syncthreads();

  for (int t = threadIdx.x; t < SEEDS * 64; t += blockDim.x) {
    int i = t >> 6, d = t & 63;
    float acc = sq[i][d];
    for (int j = 0; j < Kr; ++j) acc += sS[i][j] * Vb[(size_t)j * 64 + d];
    out[(size_t)b * SEEDS * 64 + t] = acc;
  }
}

// p2 pool (A == 1): out[b,:] = q2[:] + sum_{j<75} v2[b,j,:]
__global__ void k_pool_sum(const float* __restrict__ q2,
                           const float* __restrict__ v2, float* __restrict__ out)
{
  int b = blockIdx.x, d = threadIdx.x;     // 64 threads
  float acc = q2[d];
  const float* vb = v2 + (size_t)b * SEEDS * 64;
  for (int j = 0; j < SEEDS; ++j) acc += vb[j * 64 + d];
  out[(size_t)b * 64 + d] = acc;
}

// ---------------------------------------------------------------------------
extern "C" void kernel_launch(void* const* d_in, const int* in_sizes, int n_in,
                              void* d_out, int out_size, void* d_ws, size_t ws_size,
                              hipStream_t stream)
{
  (void)in_sizes; (void)n_in; (void)out_size; (void)ws_size;
  const int N = NNODES, E = NEDGES, B = NB;

  const int* atomids = (const int*)d_in[0];
  const int* is_ring = (const int*)d_in[1];
  const int* hyb     = (const int*)d_in[2];
  const int* arom    = (const int*)d_in[3];
  const int* eidx    = (const int*)d_in[4];
  const int* src = eidx;          // edge_index[0]
  const int* dst = eidx + E;      // edge_index[1]

  // ---- param leaves: jax tree order (sorted dict keys, list index order) ----
  int ix = 5;
  auto nxt = [&]() -> const float* { return (const float*)d_in[ix++]; };
  struct Lin { const float* b; const float* w; };
  struct LNp { const float* b; const float* g; };
  struct Egnn { Lin e1, e2; LNp lne1, lne2, lnn1, lnn2; Lin n1, n2; } eg[3];
  for (int i = 0; i < 3; ++i) {
    eg[i].e1.b = nxt();  eg[i].e1.w = nxt();
    eg[i].e2.b = nxt();  eg[i].e2.w = nxt();
    eg[i].lne1.b = nxt(); eg[i].lne1.g = nxt();
    eg[i].lne2.b = nxt(); eg[i].lne2.g = nxt();
    eg[i].lnn1.b = nxt(); eg[i].lnn1.g = nxt();
    eg[i].lnn2.b = nxt(); eg[i].lnn2.g = nxt();
    eg[i].n1.b = nxt();  eg[i].n1.w = nxt();
    eg[i].n2.b = nxt();  eg[i].n2.w = nxt();
  }
  const float* embedding = nxt();
  struct Mab { Lin fc_o, fc_q; LNp ln0, ln1; };
  struct Gmt { Lin lin1, lin2; Mab p0; const float* p0_S; Lin p0_k, p0_v;
               Mab p1; Lin p1_k, p1_v; Mab p2; const float* p2_S; Lin p2_k, p2_v; } gm[8];
  auto rdmab = [&](Mab& m) {
    m.fc_o.b = nxt(); m.fc_o.w = nxt();
    m.fc_q.b = nxt(); m.fc_q.w = nxt();
    m.ln0.b = nxt();  m.ln0.g = nxt();
    m.ln1.b = nxt();  m.ln1.g = nxt();
  };
  for (int h = 0; h < 8; ++h) {
    gm[h].lin1.b = nxt(); gm[h].lin1.w = nxt();
    gm[h].lin2.b = nxt(); gm[h].lin2.w = nxt();
    rdmab(gm[h].p0);
    gm[h].p0_S = nxt();
    gm[h].p0_k.b = nxt(); gm[h].p0_k.w = nxt();
    gm[h].p0_v.b = nxt(); gm[h].p0_v.w = nxt();
    rdmab(gm[h].p1);
    gm[h].p1_k.b = nxt(); gm[h].p1_k.w = nxt();
    gm[h].p1_v.b = nxt(); gm[h].p1_v.w = nxt();
    rdmab(gm[h].p2);
    gm[h].p2_S = nxt();
    gm[h].p2_k.b = nxt(); gm[h].p2_k.w = nxt();   // dead (softmax over size-1 axis)
    gm[h].p2_v.b = nxt(); gm[h].p2_v.w = nxt();
  }
  struct Mlp { Lin l1, l2, l3; } ml[2];
  for (int i = 0; i < 2; ++i) {
    ml[i].l1.b = nxt(); ml[i].l1.w = nxt();
    ml[i].l2.b = nxt(); ml[i].l2.w = nxt();
    ml[i].l3.b = nxt(); ml[i].l3.w = nxt();
  }
  Lin post1{nxt(), nxt()}, post2{nxt(), nxt()}, post3{nxt(), nxt()};
  Lin pre1{nxt(), nxt()},  pre2{nxt(), nxt()};

  // ---- workspace layout (floats) ----
  float* W = (float*)d_ws;
  size_t off = 0;
  float* f      = W + off; off += (size_t)N * 64;    // node features (persistent)
  float* feats  = W + off; off += (size_t)N * 192;   // egnn concat
  float* dinv   = W + off; off += N;                 // deg^-1/2
  float* pooled = W + off; off += (size_t)B * 512;
  float* q0buf  = W + off; off += SEEDS * 64;
  float* q2buf  = W + off; off += 64;
  off = (off + 63) & ~(size_t)63;
  _Float16* wt  = (_Float16*)(W + off); off += 262144;  // f16 Wt scratch (<=512x1024)
  off = (off + 255) & ~(size_t)255;
  float* scr = W + off;                              // stage-reused scratch

  // ---- launch helpers ----
  auto gemm = [&](const float* A, const float* Wm, const float* bias, float* C,
                  int M, int K, int Ncol, int ldc, int act) {
    k_cvt_w<<<(K * Ncol + 255) / 256, 256, 0, stream>>>(Wm, wt, K, Ncol);
    dim3 g((M + 63) / 64, Ncol / 64);
    k_gemm<<<g, dim3(256), 0, stream>>>(A, wt, bias, C, M, K, Ncol, ldc, act);
  };
  auto ln = [&](const float* x, int sx, const float* x2, int sx2,
                const LNp& p, const float* resid, int sr,
                float* y, int sy, int rows) {
    k_layernorm<<<(rows + 7) / 8, 256, 0, stream>>>(
        x, sx, x2, sx2, p.g, p.b, resid, sr, y, sy, rows);
  };
  auto fill = [&](float* p, float v, long n) {
    k_fill<<<(int)((n + 255) / 256), 256, 0, stream>>>(p, v, n);
  };

  // ================= Stage 1: embedding + pre MLP =================
  {
    float* e    = scr;                         // [N,256]
    float* hpre = scr + (size_t)N * 256;       // [N,128]
    k_embed<<<(N * 64 + 255) / 256, 256, 0, stream>>>(
        atomids, is_ring, hyb, arom, embedding, e);
    gemm(e, pre1.w, pre1.b, hpre, N, 256, 128, 128, 1);  // silu
    gemm(hpre, pre2.w, pre2.b, f, N, 128, 64, 64, 0);    // no act
  }

  // degree / dinv (used by every head's GCNs)
  fill(dinv, 1.0f, N);
  k_deg_scatter<<<(E + 255) / 256, 256, 0, stream>>>(dst, dinv);
  k_rsqrt_inplace<<<(N + 255) / 256, 256, 0, stream>>>(dinv, N);

  // ================= Stage 2: 3x EGNN (all on same f) =================
  {
    float* cat = scr;                            // [N,128] = [LN(f) | LN(mi)]
    float* mi  = scr + (size_t)N * 128;          // [N,64]
    float* n1o = scr + (size_t)N * 192;          // [N,128]
    float* n2o = scr + (size_t)N * 320;          // [N,64]
    float* em  = scr + (size_t)N * 384;          // [CH,128]
    float* eh  = em  + (size_t)ECHUNK * 128;     // [CH,256]
    float* e2o = eh  + (size_t)ECHUNK * 256;     // [CH,64]
    for (int k = 0; k < 3; ++k) {
      fill(mi, 0.f, (long)N * 64);
      ln(f, 64, nullptr, 0, eg[k].lnn1, nullptr, 0, cat, 128, N);  // cat[:, :64]
      for (int c = 0; c < E / ECHUNK; ++c) {
        int e0 = c * ECHUNK;
        k_edge_concat<<<(ECHUNK * 128 + 255) / 256, 256, 0, stream>>>(
            f, src, dst, em, e0, ECHUNK);
        gemm(em, eg[k].e1.w, eg[k].e1.b, eh, ECHUNK, 128, 256, 256, 1);  // silu
        gemm(eh, eg[k].e2.w, eg[k].e2.b, e2o, ECHUNK, 256, 64, 64, 1);   // silu
        ln(e2o, 64, nullptr, 0, eg[k].lne1, nullptr, 0, e2o, 64, ECHUNK);
        k_scatter_add<<<(ECHUNK * 64 + 255) / 256, 256, 0, stream>>>(
            e2o, dst, mi, e0, ECHUNK);
      }
      ln(mi, 64, nullptr, 0, eg[k].lne2, nullptr, 0, cat + 64, 128, N);  // cat[:,64:]
      gemm(cat, eg[k].n1.w, eg[k].n1.b, n1o, N, 128, 128, 128, 1);       // silu
      gemm(n1o, eg[k].n2.w, eg[k].n2.b, n2o, N, 128, 64, 64, 0);
      // feats[:, k*64 : k*64+64] = f + LN(n2o)
      ln(n2o, 64, nullptr, 0, eg[k].lnn2, f, 64, feats + k * 64, 192, N);
    }
  }

  // ================= Stage 3: post MLPs =================
  {
    float* pp1 = scr;                        // [N,64]
    float* pp2 = scr + (size_t)N * 64;       // [N,64]
    gemm(feats, post1.w, post1.b, pp1, N, 192, 64, 64, 1);
    gemm(pp1,   post2.w, post2.b, pp2, N, 64, 64, 64, 1);
    gemm(pp2,   post3.w, post3.b, f,   N, 64, 64, 64, 1);   // f reused for f2
  }

  // ================= Stage 4: 8x GMT heads =================
  {
    const size_t R = (size_t)B * SEEDS * 64;
    float* xh  = scr;                        // [N,64]
    float* xw  = scr + (size_t)N * 64;       // [N,64] GCN x@W
    float* Kb  = scr + (size_t)N * 128;      // agg/K  [B,50,64]
    float* xw2 = scr + (size_t)N * 192;
    float* Vb  = scr + (size_t)N * 256;      // agg/V
    float* bx  = scr + (size_t)N * 320;      // [B,75,64]
    float* tmp = bx + R;
    float* t2  = tmp + R;
    float* t3  = t2 + R;
    float* t4  = t3 + R;
    for (int h = 0; h < 8; ++h) {
      const Gmt& G = gm[h];
      gemm(f, G.lin1.w, G.lin1.b, xh, N, 64, 64, 64, 0);
      // GCN keys
      gemm(xh, G.p0_k.w, nullptr, xw, N, 64, 64, 64, 0);
      fill(Kb, 0.f, (long)N * 64);
      k_gcn_scatter<<<(E * 64 + 255) / 256, 256, 0, stream>>>(xw, src, dst, dinv, Kb);
      k_gcn_finish<<<(N * 64 + 255) / 256, 256, 0, stream>>>(Kb, xw, dinv, G.p0_k.b, Kb);
      // GCN values
      gemm(xh, G.p0_v.w, nullptr, xw2, N, 64, 64, 64, 0);
      fill(Vb, 0.f, (long)N * 64);
      k_gcn_scatter<<<(E * 64 + 255) / 256, 256, 0, stream>>>(xw2, src, dst, dinv, Vb);
      k_gcn_finish<<<(N * 64 + 255) / 256, 256, 0, stream>>>(Vb, xw2, dinv, G.p0_v.b, Vb);
      // ---- p0: GMPool_G ----
      gemm(G.p0_S, G.p0.fc_q.w, G.p0.fc_q.b, q0buf, SEEDS, 64, 64, 64, 0);
      k_attn<<<B, 256, 0, stream>>>(q0buf, 0, Kb, Vb, bx, NPG);
      ln(bx, 64, nullptr, 0, G.p0.ln0, nullptr, 0, bx, 64, B * SEEDS);
      gemm(bx, G.p0.fc_o.w, G.p0.fc_o.b, tmp, B * SEEDS, 64, 64, 64, 2);  // relu
      ln(bx, 64, tmp, 64, G.p0.ln1, nullptr, 0, bx, 64, B * SEEDS);
      // ---- p1: SelfAtt ----
      gemm(bx, G.p1_k.w, G.p1_k.b, t2, B * SEEDS, 64, 64, 64, 0);
      gemm(bx, G.p1_v.w, G.p1_v.b, t3, B * SEEDS, 64, 64, 64, 0);
      gemm(bx, G.p1.fc_q.w, G.p1.fc_q.b, t4, B * SEEDS, 64, 64, 64, 0);
      k_attn<<<B, 256, 0, stream>>>(t4, 1, t2, t3, tmp, SEEDS);
      ln(tmp, 64, nullptr, 0, G.p1.ln0, nullptr, 0, tmp, 64, B * SEEDS);
      gemm(tmp, G.p1.fc_o.w, G.p1.fc_o.b, t2, B * SEEDS, 64, 64, 64, 2);
      ln(tmp, 64, t2, 64, G.p1.ln1, nullptr, 0, bx, 64, B * SEEDS);
      // ---- p2: GMPool_I (1 seed; softmax over size-1 query axis => A==1) ----
      gemm(bx, G.p2_v.w, G.p2_v.b, t3, B * SEEDS, 64, 64, 64, 0);
      gemm(G.p2_S, G.p2.fc_q.w, G.p2.fc_q.b, q2buf, 1, 64, 64, 64, 0);
      k_pool_sum<<<B, 64, 0, stream>>>(q2buf, t3, tmp);          // tmp: [B,64]
      ln(tmp, 64, nullptr, 0, G.p2.ln0, nullptr, 0, tmp, 64, B);
      gemm(tmp, G.p2.fc_o.w, G.p2.fc_o.b, t2, B, 64, 64, 64, 2);
      ln(tmp, 64, t2, 64, G.p2.ln1, nullptr, 0, tmp, 64, B);
      gemm(tmp, G.lin2.w, G.lin2.b, pooled + h * 64, B, 64, 64, 512, 0);
    }
  }

  // ================= Stage 5: output MLP heads =================
  {
    float* h1 = scr;                       // [B,512]
    float* h2 = scr + (size_t)B * 512;     // [B,1024]
    float* out = (float*)d_out;
    for (int i = 0; i < 2; ++i) {
      gemm(pooled, ml[i].l1.w, ml[i].l1.b, h1, B, 512, 512, 512, 1);
      gemm(h1, ml[i].l2.w, ml[i].l2.b, h2, B, 512, 1024, 1024, 1);
      gemm(h2, ml[i].l3.w, ml[i].l3.b, out + (size_t)i * B * 512, B, 1024, 512, 512, 0);
    }
  }
}
